// Transformer_40140764348599
// MI455X (gfx1250) — matrix operand order repaired and checked by reading, not compiled, not run on hardware
//
#include <hip/hip_runtime.h>
#include <math.h>

#define NN   16000
#define NE   256000
#define DIM  256
#define HEADS 8
#define DH   32
#define FF   1024
#define BLOCKS 3

typedef __attribute__((ext_vector_type(16))) __bf16 v16bf;
typedef __attribute__((ext_vector_type(8)))  float  v8f;
typedef __attribute__((ext_vector_type(4)))  int    int4v;

// ---- gfx1250 async global->LDS path (guarded; falls back to ds staging) ----
#if defined(__AMDGCN__) && __has_builtin(__builtin_amdgcn_global_load_async_to_lds_b128)
#define HAVE_ASYNC_LDS 1
#else
#define HAVE_ASYNC_LDS 0
#endif

#if HAVE_ASYNC_LDS
static __device__ __forceinline__ void async_cp16(const void* g, void* l) {
    typedef __attribute__((address_space(1))) int4v* gp_t;
    typedef __attribute__((address_space(3))) int4v* lp_t;
    gp_t gp = (gp_t)(__attribute__((address_space(1))) void*)(void*)g;
    lp_t lp = (lp_t)(__attribute__((address_space(3))) void*)l;
    __builtin_amdgcn_global_load_async_to_lds_b128(gp, lp, 0, 0);
}
#if __has_builtin(__builtin_amdgcn_s_wait_asynccnt)
#define WAIT_ASYNC(n) __builtin_amdgcn_s_wait_asynccnt(n)
#else
#define WAIT_ASYNC(n) asm volatile("s_wait_asynccnt %0" ::"i"(n))
#endif
#else
#define WAIT_ASYNC(n)
#endif

static __device__ __forceinline__ unsigned short f32_to_bf16(float f) {
    unsigned int u = __float_as_uint(f);
    u += 0x7FFFu + ((u >> 16) & 1u);          // round-to-nearest-even
    return (unsigned short)(u >> 16);
}

// ---------------------------------------------------------------- conversions
__global__ __launch_bounds__(256) void cvt_f32_bf16(const float* __restrict__ in,
                                                    unsigned short* __restrict__ out,
                                                    int n) {
    int i = blockIdx.x * 256 + threadIdx.x;
    if (i < n) out[i] = f32_to_bf16(in[i]);
}

// out[n*K + k] = bf16(in[k*N + n])   (weights K x N  ->  bf16 N x K)
__global__ __launch_bounds__(256) void cvt_f32_bf16_T(const float* __restrict__ in,
                                                      unsigned short* __restrict__ out,
                                                      int Kd, int Nd) {
    int i = blockIdx.x * 256 + threadIdx.x;
    if (i >= Kd * Nd) return;
    int k = i / Nd, n = i - k * Nd;
    out[(size_t)n * Kd + k] = f32_to_bf16(in[i]);
}

__global__ __launch_bounds__(256) void fill_f32(float* __restrict__ p, float v, int n) {
    int i = blockIdx.x * 256 + threadIdx.x;
    if (i < n) p[i] = v;
}

// ---------------------------------------------------------------- WMMA GEMM
// C[M,N] = A[M,K] @ W + bias, W given PRE-TRANSPOSED as Wt[N,K] (bf16).
// M%128==0, N%64==0, K%32==0 (true for every call site).
// 256 threads = 8 waves. Block tile 128x64, per-wave tile 32x32
// (4 x v_wmma_f32_16x16x32_bf16 per K-step). LDS double-buffered; tiles are
// staged with gfx1250 async global->LDS copies when available.
__global__ __launch_bounds__(256) void gemm_bf16_wmma(
    const unsigned short* __restrict__ A,
    const unsigned short* __restrict__ Wt,
    const float* __restrict__ bias,
    float* __restrict__ C,
    int M, int N, int K, int relu)
{
    __shared__ __align__(16) unsigned short As[2][128 * 32]; // [row][k]
    __shared__ __align__(16) unsigned short Bs[2][64 * 32];  // [col][k]

    const int t       = threadIdx.x;
    const int rowBase = blockIdx.y * 128;
    const int colBase = blockIdx.x * 64;
    const int wave    = t >> 5;
    const int lane    = t & 31;
    const int half    = lane >> 4;
    const int lid     = lane & 15;
    const int wr      = wave >> 1;     // 0..3 : 32-row strip
    const int wc      = wave & 1;      // 0..1 : 32-col strip

    v8f a00 = {}, a01 = {}, a10 = {}, a11 = {};
    union Frag { uint4 q[2]; v16bf v; };

    // stage one 128x32 A tile + 64x32 B tile into LDS buffer `buf`
    auto stage = [&](int buf, int kt) {
        // A: 512 x 16B chunks, 2 per thread
        #pragma unroll
        for (int j = 0; j < 2; ++j) {
            int ch = t + j * 256;
            int r = ch >> 2, c = (ch & 3) << 3;
            const unsigned short* g = A + (size_t)(rowBase + r) * K + kt + c;
            unsigned short* l = &As[buf][r * 32 + c];
#if HAVE_ASYNC_LDS
            async_cp16(g, l);
#else
            *(uint4*)l = *(const uint4*)g;
#endif
        }
        // B: 256 x 16B chunks, 1 per thread
        {
            int n = t >> 2, c = (t & 3) << 3;
            const unsigned short* g = Wt + (size_t)(colBase + n) * K + kt + c;
            unsigned short* l = &Bs[buf][n * 32 + c];
#if HAVE_ASYNC_LDS
            async_cp16(g, l);
#else
            *(uint4*)l = *(const uint4*)g;
            if (kt + 32 < K)   // keep the gfx1250 prefetch path in fallback mode
                __builtin_prefetch(A + (size_t)(rowBase + (t >> 2)) * K + kt + 32, 0, 1);
#endif
        }
    };

    stage(0, 0);
    int buf = 0;
    for (int kt = 0; kt < K; kt += 32, buf ^= 1) {
        const bool hasNext = (kt + 32) < K;
        if (hasNext) stage(buf ^ 1, kt + 32);
        // current stage complete when only the just-issued 3 async ops remain
        if (hasNext) { WAIT_ASYNC(3); } else { WAIT_ASYNC(0); }
        __syncthreads();

        // per-lane fragments (CDNA5 wave32 16-bit layouts):
        // A lanes 0-15: K={0..7,16..23}; lanes 16-31: K={8..15,24..31}
        // B lanes 0-15: K={0..7,16..23}; lanes 16-31: K={8..15,24..31}
        Frag a0, a1, b0, b1;
        {
            const unsigned short* p = &As[buf][(wr * 32 + lid) * 32];
            a0.q[0] = *(const uint4*)(p + 8 * half);
            a0.q[1] = *(const uint4*)(p + 16 + 8 * half);
            const unsigned short* p1 = &As[buf][(wr * 32 + 16 + lid) * 32];
            a1.q[0] = *(const uint4*)(p1 + 8 * half);
            a1.q[1] = *(const uint4*)(p1 + 16 + 8 * half);
        }
        {
            const unsigned short* p = &Bs[buf][(wc * 32 + lid) * 32 + 8 * half];
            b0.q[0] = *(const uint4*)(p);
            b0.q[1] = *(const uint4*)(p + 16);
            const unsigned short* p1 = &Bs[buf][(wc * 32 + 16 + lid) * 32 + 8 * half];
            b1.q[0] = *(const uint4*)(p1);
            b1.q[1] = *(const uint4*)(p1 + 16);
        }

        a00 = __builtin_amdgcn_wmma_f32_16x16x32_bf16(false, a0.v, false, b0.v, (short)0, a00, false, false);
        a01 = __builtin_amdgcn_wmma_f32_16x16x32_bf16(false, a0.v, false, b1.v, (short)0, a01, false, false);
        a10 = __builtin_amdgcn_wmma_f32_16x16x32_bf16(false, a1.v, false, b0.v, (short)0, a10, false, false);
        a11 = __builtin_amdgcn_wmma_f32_16x16x32_bf16(false, a1.v, false, b1.v, (short)0, a11, false, false);
        __syncthreads();   // all waves done reading buf before it is restaged
    }

    // epilogue: C layout = VGPR r -> row r + 8*half, col = lid
    const int cn0 = colBase + wc * 32 + lid;
    const int cn1 = cn0 + 16;
    const float bv0 = bias ? bias[cn0] : 0.0f;
    const float bv1 = bias ? bias[cn1] : 0.0f;
    const int rm0 = rowBase + wr * 32 + 8 * half;
    const int rm1 = rm0 + 16;
    #pragma unroll
    for (int r = 0; r < 8; ++r) {
        float v00 = a00[r] + bv0, v01 = a01[r] + bv1;
        float v10 = a10[r] + bv0, v11 = a11[r] + bv1;
        if (relu) {
            v00 = fmaxf(v00, 0.0f); v01 = fmaxf(v01, 0.0f);
            v10 = fmaxf(v10, 0.0f); v11 = fmaxf(v11, 0.0f);
        }
        C[(size_t)(rm0 + r) * N + cn0] = v00;
        C[(size_t)(rm0 + r) * N + cn1] = v01;
        C[(size_t)(rm1 + r) * N + cn0] = v10;
        C[(size_t)(rm1 + r) * N + cn1] = v11;
    }
}

// ---------------------------------------------------------------- attention
__global__ __launch_bounds__(256) void edge_logits(
    const int* __restrict__ src, const int* __restrict__ dst,
    const float* __restrict__ Q, const float* __restrict__ Knode,
    const float* __restrict__ Ep, const float* __restrict__ dist,
    float* __restrict__ alpha, float* __restrict__ amax)
{
    const int e    = blockIdx.x * 8 + (threadIdx.x >> 5);
    const int lane = threadIdx.x & 31;
    const int s = src[e], d = dst[e];
    const float* qp = Q     + (size_t)d * DIM + lane * 8;
    const float* kp = Knode + (size_t)s * DIM + lane * 8;
    const float* ep = Ep    + (size_t)e * DIM + lane * 8;
    float p = 0.0f;
    #pragma unroll
    for (int j = 0; j < 8; ++j) p += qp[j] * (kp[j] + ep[j]);
    p += __shfl_xor(p, 1, 32);
    p += __shfl_xor(p, 2, 32);
    if ((lane & 3) == 0) {
        const int h = lane >> 2;
        const float a = p * (1.0f / dist[e]) * 0.17677669529663687f; // 1/sqrt(32)
        alpha[(size_t)e * HEADS + h] = a;
        float* addr = amax + (size_t)d * HEADS + h;
        if (a >= 0.0f) atomicMax((int*)addr, __float_as_int(a));
        else           atomicMin((unsigned int*)addr, __float_as_uint(a));
    }
}

__global__ __launch_bounds__(256) void edge_exp(
    const int* __restrict__ dst, const float* __restrict__ alpha,
    const float* __restrict__ amax, float* __restrict__ ex,
    float* __restrict__ denom)
{
    const int idx = blockIdx.x * 256 + threadIdx.x;
    if (idx >= NE * HEADS) return;
    const int e = idx >> 3, h = idx & 7;
    const int d = dst[e];
    const float v = __expf(alpha[idx] - amax[(size_t)d * HEADS + h]);
    ex[idx] = v;
    atomicAdd(denom + (size_t)d * HEADS + h, v);
}

__global__ __launch_bounds__(256) void edge_aggregate(
    const int* __restrict__ src, const int* __restrict__ dst,
    const float* __restrict__ V, const float* __restrict__ Ep,
    const float* __restrict__ ex, const float* __restrict__ denom,
    float* __restrict__ agg)
{
    const int e    = blockIdx.x * 8 + (threadIdx.x >> 5);
    const int lane = threadIdx.x & 31;
    const int s = src[e], d = dst[e];
    const int h = lane >> 2;
    const float attn = ex[(size_t)e * HEADS + h] /
                       (denom[(size_t)d * HEADS + h] + 1e-16f);
    const float* vp = V  + (size_t)s * DIM + lane * 8;
    const float* ep = Ep + (size_t)e * DIM + lane * 8;
    float* op = agg + (size_t)d * DIM + lane * 8;
    #pragma unroll
    for (int j = 0; j < 8; ++j) atomicAdd(op + j, (vp[j] + ep[j]) * attn);
}

// ---------------------------------------------------------------- elementwise
__global__ __launch_bounds__(256) void add3(const float* __restrict__ a,
                                            const float* __restrict__ b,
                                            const float* __restrict__ c,
                                            float* __restrict__ o, int n) {
    int i = blockIdx.x * 256 + threadIdx.x;
    if (i < n) o[i] = a[i] + b[i] + c[i];
}

__global__ __launch_bounds__(256) void add2(const float* __restrict__ a,
                                            const float* __restrict__ b,
                                            float* __restrict__ o, int n) {
    int i = blockIdx.x * 256 + threadIdx.x;
    if (i < n) o[i] = a[i] + b[i];
}

// ---------------------------------------------------------------- graph LN
__global__ __launch_bounds__(256) void reduce_stats(const float* __restrict__ y,
                                                    float* __restrict__ sums, int n) {
    __shared__ float s1[256], s2[256];
    const int base = blockIdx.x * 256 * 8 + threadIdx.x;
    float a = 0.0f, b = 0.0f;
    #pragma unroll
    for (int j = 0; j < 8; ++j) {
        int idx = base + j * 256;
        if (idx < n) { float v = y[idx]; a += v; b += v * v; }
    }
    s1[threadIdx.x] = a; s2[threadIdx.x] = b;
    __syncthreads();
    for (int off = 128; off > 0; off >>= 1) {
        if (threadIdx.x < off) {
            s1[threadIdx.x] += s1[threadIdx.x + off];
            s2[threadIdx.x] += s2[threadIdx.x + off];
        }
        __syncthreads();
    }
    if (threadIdx.x == 0) {
        atomicAdd(&sums[0], s1[0]);
        atomicAdd(&sums[1], s2[0]);
    }
}

__global__ __launch_bounds__(256) void ln_apply(
    const float* __restrict__ y, const float* __restrict__ sums,
    const float* __restrict__ g, const float* __restrict__ b,
    float* __restrict__ out_f32, unsigned short* __restrict__ out_bf16, int n)
{
    int i = blockIdx.x * 256 + threadIdx.x;
    if (i >= n) return;
    const float mean = sums[0] / (float)n;
    const float var  = fmaxf(sums[1] / (float)n - mean * mean, 0.0f);
    const float inv  = 1.0f / (sqrtf(var) + 1e-5f);
    const int c = i & (DIM - 1);
    const float v = (y[i] - mean) * inv * g[c] + b[c];
    out_f32[i]  = v;
    out_bf16[i] = f32_to_bf16(v);
}

// ---------------------------------------------------------------- launch
extern "C" void kernel_launch(void* const* d_in, const int* in_sizes, int n_in,
                              void* d_out, int out_size, void* d_ws, size_t ws_size,
                              hipStream_t stream)
{
    const float* x      = (const float*)d_in[0];
    const int*   eidx   = (const int*)  d_in[1];
    const float* xedge  = (const float*)d_in[2];
    const float* xdist  = (const float*)d_in[3];
    const float* Wq = (const float*)d_in[4];   const float* bq = (const float*)d_in[5];
    const float* Wk = (const float*)d_in[6];   const float* bk = (const float*)d_in[7];
    const float* Wv = (const float*)d_in[8];   const float* bv = (const float*)d_in[9];
    const float* We = (const float*)d_in[10];
    const float* Ws = (const float*)d_in[11];  const float* bs = (const float*)d_in[12];
    const float* g1 = (const float*)d_in[13];  const float* b1 = (const float*)d_in[14];
    const float* W1 = (const float*)d_in[15];  const float* c1 = (const float*)d_in[16];
    const float* W2 = (const float*)d_in[17];  const float* c2 = (const float*)d_in[18];
    const float* g2 = (const float*)d_in[19];  const float* b2 = (const float*)d_in[20];
    const int* srcv = eidx;
    const int* dstv = eidx + NE;

    // ---- workspace carve-up
    char* p = (char*)d_ws;
    auto alloc = [&](size_t bytes) -> char* {
        char* r = p; p += (bytes + 255) & ~(size_t)255; return r;
    };
    unsigned short* xb   = (unsigned short*)alloc((size_t)NN * DIM * 2);
    unsigned short* eb   = (unsigned short*)alloc((size_t)NE * DIM * 2);
    unsigned short* xlnb = (unsigned short*)alloc((size_t)NN * DIM * 2);
    unsigned short* Hb   = (unsigned short*)alloc((size_t)NN * FF  * 2);
    unsigned short* wqb  = (unsigned short*)alloc((size_t)DIM * DIM * 2);
    unsigned short* wkb  = (unsigned short*)alloc((size_t)DIM * DIM * 2);
    unsigned short* wvb  = (unsigned short*)alloc((size_t)DIM * DIM * 2);
    unsigned short* web  = (unsigned short*)alloc((size_t)DIM * DIM * 2);
    unsigned short* wsb  = (unsigned short*)alloc((size_t)DIM * DIM * 2);
    unsigned short* w1b  = (unsigned short*)alloc((size_t)DIM * FF  * 2);
    unsigned short* w2b  = (unsigned short*)alloc((size_t)FF  * DIM * 2);
    float* xcur  = (float*)alloc((size_t)NN * DIM * 4);
    float* y     = (float*)alloc((size_t)NN * DIM * 4);
    float* xln   = (float*)alloc((size_t)NN * DIM * 4);
    float* Qn    = (float*)alloc((size_t)NN * DIM * 4);
    float* Kn    = (float*)alloc((size_t)NN * DIM * 4);
    float* Vn    = (float*)alloc((size_t)NN * DIM * 4);
    float* Sn    = (float*)alloc((size_t)NN * DIM * 4);
    float* agg   = (float*)alloc((size_t)NN * DIM * 4);
    float* Fb    = (float*)alloc((size_t)NN * DIM * 4);
    float* Hm    = (float*)alloc((size_t)NN * FF  * 4);
    float* Ep    = (float*)alloc((size_t)NE * DIM * 4);
    float* alpha = (float*)alloc((size_t)NE * HEADS * 4);
    float* ex    = (float*)alloc((size_t)NE * HEADS * 4);
    float* amax  = (float*)alloc((size_t)NN * HEADS * 4);
    float* denom = (float*)alloc((size_t)NN * HEADS * 4);
    float* sums  = (float*)alloc(256);

    const int nND = NN * DIM;          // 4,096,000
    const int nED = NE * DIM;          // 65,536,000 (fits int)
    dim3 blk(256);
    auto g1d = [](int n) { return dim3((unsigned)((n + 255) / 256)); };

    cvt_f32_bf16<<<g1d(nND), blk, 0, stream>>>(x, xb, nND);
    cvt_f32_bf16<<<g1d(nED), blk, 0, stream>>>(xedge, eb, nED);
    (void)hipMemcpyAsync(xcur, x, (size_t)nND * 4, hipMemcpyDeviceToDevice, stream);

    for (int i = 0; i < BLOCKS; ++i) {
        const float* Wq_i = Wq + (size_t)i * DIM * DIM;
        const float* Wk_i = Wk + (size_t)i * DIM * DIM;
        const float* Wv_i = Wv + (size_t)i * DIM * DIM;
        const float* We_i = We + (size_t)i * DIM * DIM;
        const float* Ws_i = Ws + (size_t)i * DIM * DIM;
        const float* W1_i = W1 + (size_t)i * DIM * FF;
        const float* W2_i = W2 + (size_t)i * FF * DIM;

        // weights -> bf16, pre-transposed to [N][K] for straight-row staging
        cvt_f32_bf16_T<<<g1d(DIM*DIM), blk, 0, stream>>>(Wq_i, wqb, DIM, DIM);
        cvt_f32_bf16_T<<<g1d(DIM*DIM), blk, 0, stream>>>(Wk_i, wkb, DIM, DIM);
        cvt_f32_bf16_T<<<g1d(DIM*DIM), blk, 0, stream>>>(Wv_i, wvb, DIM, DIM);
        cvt_f32_bf16_T<<<g1d(DIM*DIM), blk, 0, stream>>>(We_i, web, DIM, DIM);
        cvt_f32_bf16_T<<<g1d(DIM*DIM), blk, 0, stream>>>(Ws_i, wsb, DIM, DIM);
        cvt_f32_bf16_T<<<g1d(DIM*FF),  blk, 0, stream>>>(W1_i, w1b, DIM, FF);
        cvt_f32_bf16_T<<<g1d(FF*DIM),  blk, 0, stream>>>(W2_i, w2b, FF, DIM);

        // ---- projections (WMMA GEMMs), tile 128x64
        gemm_bf16_wmma<<<dim3(DIM/64, NN/128), blk, 0, stream>>>(xb, wqb, bq + i*DIM, Qn, NN, DIM, DIM, 0);
        gemm_bf16_wmma<<<dim3(DIM/64, NN/128), blk, 0, stream>>>(xb, wkb, bk + i*DIM, Kn, NN, DIM, DIM, 0);
        gemm_bf16_wmma<<<dim3(DIM/64, NN/128), blk, 0, stream>>>(xb, wvb, bv + i*DIM, Vn, NN, DIM, DIM, 0);
        gemm_bf16_wmma<<<dim3(DIM/64, NN/128), blk, 0, stream>>>(xb, wsb, bs + i*DIM, Sn, NN, DIM, DIM, 0);
        gemm_bf16_wmma<<<dim3(DIM/64, NE/128), blk, 0, stream>>>(eb, web, nullptr,    Ep, NE, DIM, DIM, 0);

        // ---- segment softmax attention
        fill_f32<<<g1d(NN*HEADS), blk, 0, stream>>>(amax, -INFINITY, NN*HEADS);
        (void)hipMemsetAsync(denom, 0, (size_t)NN*HEADS*4, stream);
        (void)hipMemsetAsync(agg,   0, (size_t)nND*4, stream);
        edge_logits   <<<dim3(NE/8), blk, 0, stream>>>(srcv, dstv, Qn, Kn, Ep, xdist, alpha, amax);
        edge_exp      <<<g1d(NE*HEADS), blk, 0, stream>>>(dstv, alpha, amax, ex, denom);
        edge_aggregate<<<dim3(NE/8), blk, 0, stream>>>(srcv, dstv, Vn, Ep, ex, denom, agg);

        // ---- residual + graph LN 1
        add3<<<g1d(nND), blk, 0, stream>>>(xcur, agg, Sn, y, nND);
        (void)hipMemsetAsync(sums, 0, 8, stream);
        reduce_stats<<<g1d((nND + 7) / 8), blk, 0, stream>>>(y, sums, nND);
        ln_apply<<<g1d(nND), blk, 0, stream>>>(y, sums, g1 + i*DIM, b1 + i*DIM, xln, xlnb, nND);

        // ---- FFN (WMMA GEMMs) + residual + graph LN 2
        gemm_bf16_wmma<<<dim3(FF/64,  NN/128), blk, 0, stream>>>(xlnb, w1b, c1 + i*FF, Hm, NN, FF, DIM, 1);
        cvt_f32_bf16<<<g1d(NN*FF), blk, 0, stream>>>(Hm, Hb, NN*FF);
        gemm_bf16_wmma<<<dim3(DIM/64, NN/128), blk, 0, stream>>>(Hb, w2b, c2 + i*DIM, Fb, NN, DIM, FF, 0);
        add2<<<g1d(nND), blk, 0, stream>>>(xln, Fb, y, nND);
        (void)hipMemsetAsync(sums, 0, 8, stream);
        reduce_stats<<<g1d((nND + 7) / 8), blk, 0, stream>>>(y, sums, nND);
        float* outp = (i == BLOCKS - 1) ? (float*)d_out : xcur;
        ln_apply<<<g1d(nND), blk, 0, stream>>>(y, sums, g2 + i*DIM, b2 + i*DIM, outp, xb, nND);
    }
}
